// _GraphormerNodeLayer_44865228374490
// MI455X (gfx1250) — matrix operand order repaired and checked by reading, not compiled, not run on hardware
//
#include <hip/hip_runtime.h>
#include <hip/hip_bf16.h>
#include <math.h>

#define NND   50000
#define NED   800000
#define DM    128
#define NH    8
#define DH    16
#define DFF   512
#define LN_EPS 1e-5f

typedef __attribute__((ext_vector_type(16))) __bf16        v16bf;
typedef __attribute__((ext_vector_type(8)))  float         v8f;
typedef __attribute__((ext_vector_type(8)))  unsigned int  v8u;

static __device__ __forceinline__ unsigned short f2bf(float f) {
    unsigned int u = __builtin_bit_cast(unsigned int, f);
    unsigned int r = u + 0x7FFFu + ((u >> 16) & 1u);
    return (unsigned short)(r >> 16);
}

static __device__ __forceinline__ v8f do_wmma(v8u au, v8u bu, v8f acc) {
    v16bf a = __builtin_bit_cast(v16bf, au);
    v16bf b = __builtin_bit_cast(v16bf, bu);
    return __builtin_amdgcn_wmma_f32_16x16x32_bf16(false, a, false, b,
                                                   (short)0, acc, false, false);
}

// Load A fragment: 16-bit A 16x32 layout == two contiguous 16B spans of the row.
static __device__ __forceinline__ v8u load_afrag(const unsigned short* __restrict__ A,
                                                 size_t rowbase, int kt, int g) {
    const unsigned short* p = A + rowbase + kt * 32 + 8 * g;
    uint4 a0 = *(const uint4*)(p);
    uint4 a1 = *(const uint4*)(p + 16);
    v8u au;
    au[0] = a0.x; au[1] = a0.y; au[2] = a0.z; au[3] = a0.w;
    au[4] = a1.x; au[5] = a1.y; au[6] = a1.z; au[7] = a1.w;
    return au;
}

// ---------------- conversions ----------------
__global__ void k_f32_to_bf16(const float* __restrict__ src,
                              unsigned short* __restrict__ dst, int n) {
    int i = blockIdx.x * blockDim.x + threadIdx.x;
    if (i < n) dst[i] = f2bf(src[i]);
}

// ---------------- weight pre-swizzle into WMMA B-fragment order ----------------
// Output dword t = ((kt*nct + ct)*32 + lane)*8 + v packs W[kt*32+8g+2(v&3)+16(v>>2)][ct*16+n] (lo)
// and W[..+1][..] (hi), so each lane reads its whole B fragment as one 32B load.
__global__ void k_w_swizzle(const float* __restrict__ W, unsigned int* __restrict__ Wf,
                            int K, int N) {
    int t = blockIdx.x * blockDim.x + threadIdx.x;
    int total = (K / 2) * N;
    if (t >= total) return;
    int v    = t & 7;
    int lane = (t >> 3) & 31;
    int rest = t >> 8;
    int nct  = N / 16;
    int ct   = rest % nct;
    int kt   = rest / nct;
    int g = lane >> 4;
    int n = ct * 16 + (lane & 15);
    int kk0 = kt * 32 + 8 * g + 2 * (v & 3) + 16 * (v >> 2);
    unsigned int lo = f2bf(W[(size_t)kk0 * N + n]);
    unsigned int hi = f2bf(W[(size_t)(kk0 + 1) * N + n]);
    Wf[t] = lo | (hi << 16);
}

// ---------------- CSR rowptr (row is sorted) ----------------
__global__ void k_rowptr(const int* __restrict__ row, int* __restrict__ rowptr) {
    int i = blockIdx.x * blockDim.x + threadIdx.x;
    if (i > NND) return;
    int lo = 0, hi = NED;
    while (lo < hi) { int mid = (lo + hi) >> 1; if (row[mid] < i) lo = mid + 1; else hi = mid; }
    rowptr[i] = lo;
}

// ---------------- fused QKV: one A fragment feeds 3 WMMAs ----------------
__global__ void k_qkv_gemm(const unsigned short* __restrict__ xh,
                           const unsigned int* __restrict__ Wfq,
                           const unsigned int* __restrict__ Wfk,
                           const unsigned int* __restrict__ Wfv,
                           const float* __restrict__ bq, const float* __restrict__ bk,
                           const float* __restrict__ bv,
                           float* __restrict__ Q, float* __restrict__ K, float* __restrict__ V) {
    int lane = threadIdx.x & 31, wave = threadIdx.x >> 5;
    int row0 = blockIdx.x * 16, c = wave;            // 8 col tiles of 128
    int m = lane & 15, g = lane >> 4;
    size_t rowbase = (size_t)(row0 + m) * DM;
    v8f aq = {}, ak = {}, av = {};
#pragma unroll
    for (int kt = 0; kt < 4; ++kt) {                  // K = 128
        v8u au = load_afrag(xh, rowbase, kt, g);
        size_t boff = (((size_t)kt * 8 + c) * 32 + lane) * 8;
        v8u buq = *(const v8u*)(Wfq + boff);
        v8u buk = *(const v8u*)(Wfk + boff);
        v8u buv = *(const v8u*)(Wfv + boff);
        aq = do_wmma(au, buq, aq);
        ak = do_wmma(au, buk, ak);
        av = do_wmma(au, buv, av);
    }
    int col0 = c * 16;
    float cq = bq[col0 + m], ck = bk[col0 + m], cv = bv[col0 + m];
#pragma unroll
    for (int v = 0; v < 8; ++v) {
        size_t idx = (size_t)(row0 + v + 8 * g) * DM + col0 + m;
        Q[idx] = aq[v] + cq;
        K[idx] = ak[v] + ck;
        V[idx] = av[v] + cv;
    }
}

// ---------------- edge scores: wave per edge, 8 heads ----------------
__global__ void k_edge_scores(const float* __restrict__ Q, const float* __restrict__ K,
                              const int* __restrict__ row, const int* __restrict__ col,
                              const int* __restrict__ db, const float* __restrict__ emb,
                              float* __restrict__ scores) {
    int lane = threadIdx.x & 31;
    int e = blockIdx.x * (blockDim.x >> 5) + (threadIdx.x >> 5);
    if (e >= NED) return;
    int r = row[e], c = col[e];
    float4 qa = ((const float4*)(Q + (size_t)r * DM))[lane];
    float4 ka = ((const float4*)(K + (size_t)c * DM))[lane];
    float p = qa.x * ka.x + qa.y * ka.y + qa.z * ka.z + qa.w * ka.w;
    p += __shfl_xor(p, 1, 32);
    p += __shfl_xor(p, 2, 32);
    if ((lane & 3) == 0) {
        int h = lane >> 2;
        float s = p * 0.25f + emb[db[r] * NH + h] + emb[db[c] * NH + h];
        scores[(size_t)e * NH + h] = s;
    }
}

// ---------------- segment softmax: wave per node ----------------
__global__ void k_segment_softmax(const int* __restrict__ rowptr, float* __restrict__ scores) {
    int lane = threadIdx.x & 31;
    int nIdx = blockIdx.x * (blockDim.x >> 5) + (threadIdx.x >> 5);
    if (nIdx >= NND) return;
    int start = rowptr[nIdx], cnt = rowptr[nIdx + 1] - start;
    int h = lane & 7, j0 = lane >> 3;   // 4 lanes per head, stride 4 over edges
    float m = -3.4e38f;
    for (int j = j0; j < cnt; j += 4)
        m = fmaxf(m, scores[(size_t)(start + j) * NH + h]);
    m = fmaxf(m, __shfl_xor(m, 8, 32));
    m = fmaxf(m, __shfl_xor(m, 16, 32));
    float s = 0.f;
    for (int j = j0; j < cnt; j += 4)
        s += expf(scores[(size_t)(start + j) * NH + h] - m);
    s += __shfl_xor(s, 8, 32);
    s += __shfl_xor(s, 16, 32);
    float inv = 1.f / fmaxf(s, 1e-12f);
    for (int j = j0; j < cnt; j += 4) {
        size_t idx = (size_t)(start + j) * NH + h;
        scores[idx] = expf(scores[idx] - m) * inv;
    }
}

// ---------------- aggregation: wave per node, writes bf16 agg ----------------
__global__ void k_aggregate(const int* __restrict__ rowptr, const int* __restrict__ col,
                            const float* __restrict__ alpha, const float* __restrict__ V,
                            unsigned short* __restrict__ aggh) {
    int lane = threadIdx.x & 31;
    int nIdx = blockIdx.x * (blockDim.x >> 5) + (threadIdx.x >> 5);
    if (nIdx >= NND) return;
    int start = rowptr[nIdx], end = rowptr[nIdx + 1];
    int h = lane >> 2;            // lane covers dims [4*lane, 4*lane+4)
    float ax = 0.f, ay = 0.f, az = 0.f, aw = 0.f;
    for (int j = start; j < end; ++j) {
        float a = alpha[(size_t)j * NH + h];
        int c = col[j];
        float4 v4 = ((const float4*)(V + (size_t)c * DM))[lane];
        ax += a * v4.x; ay += a * v4.y; az += a * v4.z; aw += a * v4.w;
    }
    size_t base = (size_t)nIdx * DM + lane * 4;
    aggh[base + 0] = f2bf(ax); aggh[base + 1] = f2bf(ay);
    aggh[base + 2] = f2bf(az); aggh[base + 3] = f2bf(aw);
}

// ---------------- output projection + residual ----------------
__global__ void k_oproj(const unsigned short* __restrict__ aggh,
                        const unsigned int* __restrict__ Wfo,
                        const float* __restrict__ bo, const float* __restrict__ x,
                        float* __restrict__ hpre) {
    int lane = threadIdx.x & 31, wave = threadIdx.x >> 5;
    int row0 = blockIdx.x * 16, c = wave;
    int m = lane & 15, g = lane >> 4;
    size_t rowbase = (size_t)(row0 + m) * DM;
    v8f acc = {};
#pragma unroll
    for (int kt = 0; kt < 4; ++kt) {
        v8u au = load_afrag(aggh, rowbase, kt, g);
        v8u bu = *(const v8u*)(Wfo + (((size_t)kt * 8 + c) * 32 + lane) * 8);
        acc = do_wmma(au, bu, acc);
    }
    int col0 = c * 16;
    float bc = bo[col0 + m];
#pragma unroll
    for (int v = 0; v < 8; ++v) {
        size_t idx = (size_t)(row0 + v + 8 * g) * DM + col0 + m;
        hpre[idx] = acc[v] + bc + x[idx];
    }
}

// ---------------- LayerNorm: wave per row ----------------
__global__ void k_layernorm(const float* __restrict__ in, const float* __restrict__ gamma,
                            const float* __restrict__ beta, float* __restrict__ out,
                            unsigned short* __restrict__ out_bf16) {
    int lane = threadIdx.x & 31;
    int r = blockIdx.x * (blockDim.x >> 5) + (threadIdx.x >> 5);
    if (r >= NND) return;
    float4 v = ((const float4*)(in + (size_t)r * DM))[lane];
    float s = v.x + v.y + v.z + v.w;
#pragma unroll
    for (int m = 1; m < 32; m <<= 1) s += __shfl_xor(s, m, 32);
    float mu = s * (1.f / DM);
    float dx = v.x - mu, dy = v.y - mu, dz = v.z - mu, dw = v.w - mu;
    float q = dx * dx + dy * dy + dz * dz + dw * dw;
#pragma unroll
    for (int m = 1; m < 32; m <<= 1) q += __shfl_xor(q, m, 32);
    float rstd = rsqrtf(q * (1.f / DM) + LN_EPS);
    float4 gm = ((const float4*)gamma)[lane];
    float4 bt = ((const float4*)beta)[lane];
    float4 o;
    o.x = dx * rstd * gm.x + bt.x; o.y = dy * rstd * gm.y + bt.y;
    o.z = dz * rstd * gm.z + bt.z; o.w = dw * rstd * gm.w + bt.w;
    ((float4*)(out + (size_t)r * DM))[lane] = o;
    if (out_bf16) {
        size_t base = (size_t)r * DM + lane * 4;
        out_bf16[base + 0] = f2bf(o.x); out_bf16[base + 1] = f2bf(o.y);
        out_bf16[base + 2] = f2bf(o.z); out_bf16[base + 3] = f2bf(o.w);
    }
}

// ---------------- FF1: 4 col tiles per wave, A reused 4x ----------------
__global__ void k_ff1(const unsigned short* __restrict__ hh, const unsigned int* __restrict__ Wf1,
                      const float* __restrict__ b1, unsigned short* __restrict__ z) {
    int lane = threadIdx.x & 31, wave = threadIdx.x >> 5;
    int row0 = blockIdx.x * 16;
    int c0 = wave * 4;                                  // 32 col tiles of 512
    int m = lane & 15, g = lane >> 4;
    size_t rowbase = (size_t)(row0 + m) * DM;
    v8f acc[4] = {v8f{}, v8f{}, v8f{}, v8f{}};
#pragma unroll
    for (int kt = 0; kt < 4; ++kt) {                    // K = 128
        v8u au = load_afrag(hh, rowbase, kt, g);
#pragma unroll
        for (int t = 0; t < 4; ++t) {
            v8u bu = *(const v8u*)(Wf1 + (((size_t)kt * 32 + c0 + t) * 32 + lane) * 8);
            acc[t] = do_wmma(au, bu, acc[t]);
        }
    }
#pragma unroll
    for (int t = 0; t < 4; ++t) {
        int col0 = (c0 + t) * 16;
        float bc = b1[col0 + m];
#pragma unroll
        for (int v = 0; v < 8; ++v) {
            float u = acc[t][v] + bc;
            float ge = 0.5f * u * (1.f + erff(u * 0.70710678118f));   // exact gelu
            z[(size_t)(row0 + v + 8 * g) * DFF + col0 + m] = f2bf(ge);
        }
    }
}

// ---------------- FF2: z @ W2 + b2 + h -> out_pre ----------------
__global__ void k_ff2(const unsigned short* __restrict__ z, const unsigned int* __restrict__ Wf2,
                      const float* __restrict__ b2, const float* __restrict__ hres,
                      float* __restrict__ outpre) {
    int lane = threadIdx.x & 31, wave = threadIdx.x >> 5;
    int row0 = blockIdx.x * 16, c = wave;
    int m = lane & 15, g = lane >> 4;
    size_t rowbase = (size_t)(row0 + m) * DFF;
    v8f acc = {};
#pragma unroll
    for (int kt = 0; kt < 16; ++kt) {                   // K = 512
        v8u au = load_afrag(z, rowbase, kt, g);
        v8u bu = *(const v8u*)(Wf2 + (((size_t)kt * 8 + c) * 32 + lane) * 8);
        acc = do_wmma(au, bu, acc);
    }
    int col0 = c * 16;
    float bc = b2[col0 + m];
#pragma unroll
    for (int v = 0; v < 8; ++v) {
        size_t idx = (size_t)(row0 + v + 8 * g) * DM + col0 + m;
        outpre[idx] = acc[v] + bc + hres[idx];
    }
}

extern "C" void kernel_launch(void* const* d_in, const int* in_sizes, int n_in,
                              void* d_out, int out_size, void* d_ws, size_t ws_size,
                              hipStream_t stream) {
    const float* x    = (const float*)d_in[0];
    const int*   row  = (const int*)d_in[1];
    const int*   col  = (const int*)d_in[2];
    const int*   db   = (const int*)d_in[3];
    const float* Wq   = (const float*)d_in[4];  const float* bq = (const float*)d_in[5];
    const float* Wk   = (const float*)d_in[6];  const float* bk = (const float*)d_in[7];
    const float* Wv   = (const float*)d_in[8];  const float* bv = (const float*)d_in[9];
    const float* Wo   = (const float*)d_in[10]; const float* bo = (const float*)d_in[11];
    const float* emb  = (const float*)d_in[12];
    const float* ln1g = (const float*)d_in[13]; const float* ln1b = (const float*)d_in[14];
    const float* ln2g = (const float*)d_in[15]; const float* ln2b = (const float*)d_in[16];
    const float* W1   = (const float*)d_in[17]; const float* b1   = (const float*)d_in[18];
    const float* W2   = (const float*)d_in[19]; const float* b2   = (const float*)d_in[20];
    float* out = (float*)d_out;

    char* ws = (char*)d_ws;
    size_t off = 0;
    auto alloc = [&](size_t bytes) -> void* {
        void* p = ws + off;
        off += (bytes + 255) & ~(size_t)255;
        return p;
    };

    unsigned short* xh   = (unsigned short*)alloc((size_t)NND * DM * 2);
    unsigned int*   Wfq  = (unsigned int*)alloc((size_t)DM * DM * 2);
    unsigned int*   Wfk  = (unsigned int*)alloc((size_t)DM * DM * 2);
    unsigned int*   Wfv  = (unsigned int*)alloc((size_t)DM * DM * 2);
    unsigned int*   Wfo  = (unsigned int*)alloc((size_t)DM * DM * 2);
    unsigned int*   Wf1  = (unsigned int*)alloc((size_t)DM * DFF * 2);
    unsigned int*   Wf2  = (unsigned int*)alloc((size_t)DFF * DM * 2);
    int*            rptr = (int*)alloc((size_t)(NND + 1) * 4);
    unsigned short* aggh = (unsigned short*)alloc((size_t)NND * DM * 2);
    float*          hf   = (float*)alloc((size_t)NND * DM * 4);
    unsigned short* hh   = (unsigned short*)alloc((size_t)NND * DM * 2);
    float*          Qb   = (float*)alloc((size_t)NND * DM * 4);
    float*          Kb   = (float*)alloc((size_t)NND * DM * 4);
    float*          Vb   = (float*)alloc((size_t)NND * DM * 4);   // 256B-multiple
    float*          sc   = (float*)alloc((size_t)NED * NH * 4);   // directly follows Vb
    // Aliases (lifetimes disjoint, stream-ordered):
    float*          hpre   = Qb;                   // Q dead after edge scores
    float*          outpre = Kb;                   // K dead after edge scores
    unsigned short* zbuf   = (unsigned short*)Vb;  // spans Vb+sc = N*FF*2 exactly

    // conversions / weight swizzles
    k_f32_to_bf16<<<(NND * DM + 255) / 256, 256, 0, stream>>>(x, xh, NND * DM);
    k_w_swizzle<<<32,  256, 0, stream>>>(Wq, Wfq, DM, DM);     // 8192 dwords
    k_w_swizzle<<<32,  256, 0, stream>>>(Wk, Wfk, DM, DM);
    k_w_swizzle<<<32,  256, 0, stream>>>(Wv, Wfv, DM, DM);
    k_w_swizzle<<<32,  256, 0, stream>>>(Wo, Wfo, DM, DM);
    k_w_swizzle<<<128, 256, 0, stream>>>(W1, Wf1, DM, DFF);    // 32768 dwords
    k_w_swizzle<<<128, 256, 0, stream>>>(W2, Wf2, DFF, DM);

    k_rowptr<<<(NND + 1 + 255) / 256, 256, 0, stream>>>(row, rptr);

    // attention
    k_qkv_gemm<<<NND / 16, 256, 0, stream>>>(xh, Wfq, Wfk, Wfv, bq, bk, bv, Qb, Kb, Vb);
    k_edge_scores<<<NED / 8, 256, 0, stream>>>(Qb, Kb, row, col, db, emb, sc);
    k_segment_softmax<<<NND / 8, 256, 0, stream>>>(rptr, sc);
    k_aggregate<<<NND / 8, 256, 0, stream>>>(rptr, col, sc, Vb, aggh);
    k_oproj<<<NND / 16, 256, 0, stream>>>(aggh, Wfo, bo, x, hpre);
    k_layernorm<<<NND / 8, 256, 0, stream>>>(hpre, ln1g, ln1b, hf, hh);

    // FFN
    k_ff1<<<NND / 16, 256, 0, stream>>>(hh, Wf1, b1, zbuf);
    k_ff2<<<NND / 16, 256, 0, stream>>>(zbuf, Wf2, b2, hf, outpre);
    k_layernorm<<<NND / 8, 256, 0, stream>>>(outpre, ln2g, ln2b, out, (unsigned short*)nullptr);
}
